// SparseCouplingLayer_89309549953682
// MI455X (gfx1250) — matrix operand order repaired and checked by reading, not compiled, not run on hardware
//
#include <hip/hip_runtime.h>

typedef __attribute__((ext_vector_type(16))) _Float16 v16h;
typedef __attribute__((ext_vector_type(8)))  _Float16 v8h;
typedef __attribute__((ext_vector_type(8)))  float    v8f;

#define NPTS 150000
#define KNBR 27
#define CH   64
#define TILES (NPTS / 16)              // 9375 = 5^5 * 3
#define WPB   5                        // waves per block
#define CONV_BLOCKS (TILES / WPB)      // 1875, exact

// Build a 16x32 f16 A-fragment for wave32 WMMA from a contiguous row.
// Per ISA 7.12.2: lane half h holds K = h*8 + [0..7] (elems 0-7) and
// K = 16 + h*8 + [0..7] (elems 8-15).  Caller passes row + ci0 + half*8.
__device__ __forceinline__ v16h make_a(const _Float16* base) {
  v8h lo = *(const v8h*)(base);
  v8h hi = *(const v8h*)(base + 16);
  v16h a;
#pragma unroll
  for (int e = 0; e < 8; ++e) { a[e] = lo[e]; a[e + 8] = hi[e]; }
  return a;
}

// Accumulate one 16-row tile of the gathered GEMM (N x 1728) @ (1728 x 64).
// pw: pre-packed B fragments, [54 chunks][4 ntiles][32 lanes][16] f16
// (chunk stride 2048 elems, tile stride 512 elems).
// Software-pipelined: B fragments double-buffered one chunk ahead, A
// fragments (gathered rows) fetched one full neighbor (2 chunks) ahead.
__device__ __forceinline__ void sparse_accum(
    const _Float16* __restrict__ feat, const int* __restrict__ nbr,
    const _Float16* __restrict__ pw, long rowBase, int lane, v8f acc[4]) {
  const int m = lane & 15, half = lane >> 4;
  const char* featB = (const char*)feat;
  const int* nrow = nbr + (rowBase + m) * KNBR;

  // prologue: neighbor 0 A fragments + chunk 0 B fragments
  {
  }
  unsigned bo0 = (unsigned)nrow[0] * (CH * 2u);
  const _Float16* fr0 = (const _Float16*)(featB + (size_t)bo0);
  v16h a0 = make_a(fr0 + half * 8);
  v16h a1 = make_a(fr0 + 32 + half * 8);
  v16h b[4];
  {
    const _Float16* bp0 = pw + (size_t)lane * 16;
#pragma unroll
    for (int t = 0; t < 4; ++t) b[t] = *(const v16h*)(bp0 + (size_t)t * 512);
  }

#pragma unroll
  for (int k = 0; k < KNBR; ++k) {
    // prefetch next neighbor's A fragments (used at iteration k+1)
    v16h a0n, a1n;
    if (k + 1 < KNBR) {
      unsigned bon = (unsigned)nrow[k + 1] * (CH * 2u);
      const _Float16* frn = (const _Float16*)(featB + (size_t)bon);
      a0n = make_a(frn + half * 8);
      a1n = make_a(frn + 32 + half * 8);
    }
    // prefetch B for chunk 2k+1, then WMMA chunk 2k with (a0, b)
    v16h bn[4];
    {
      const _Float16* bpn = pw + (size_t)(2 * k + 1) * 2048 + (size_t)lane * 16;
#pragma unroll
      for (int t = 0; t < 4; ++t) bn[t] = *(const v16h*)(bpn + (size_t)t * 512);
    }
#pragma unroll
    for (int t = 0; t < 4; ++t)
      acc[t] = __builtin_amdgcn_wmma_f32_16x16x32_f16(
          false, a0, false, b[t], (short)0, acc[t], false, false);
    // prefetch B for chunk 2k+2, then WMMA chunk 2k+1 with (a1, bn)
    if (k + 1 < KNBR) {
      const _Float16* bp2 = pw + (size_t)(2 * k + 2) * 2048 + (size_t)lane * 16;
#pragma unroll
      for (int t = 0; t < 4; ++t) b[t] = *(const v16h*)(bp2 + (size_t)t * 512);
    }
#pragma unroll
    for (int t = 0; t < 4; ++t)
      acc[t] = __builtin_amdgcn_wmma_f32_16x16x32_f16(
          false, a1, false, bn[t], (short)0, acc[t], false, false);
    if (k + 1 < KNBR) { a0 = a0n; a1 = a1n; }
  }
}

// Stage 1: t2 = lrelu( lrelu(sparse_conv(feat, W1)) @ W2 ), written as f16.
__global__ __launch_bounds__(WPB * 32) void k_sparse_dense(
    const _Float16* __restrict__ feat, const int* __restrict__ nbr,
    const _Float16* __restrict__ pw1, const _Float16* __restrict__ pw2,
    _Float16* __restrict__ out) {
  __shared__ __attribute__((aligned(16))) _Float16 lds[WPB][16 * CH];
  const int tid = threadIdx.x;
  const int wave = tid >> 5, lane = tid & 31;
  const int m = lane & 15, half = lane >> 4;
  const long rowBase = ((long)blockIdx.x * WPB + wave) * 16;

  v8f acc[4] = {};
  sparse_accum(feat, nbr, pw1, rowBase, lane, acc);

  // lrelu, convert to f16, scatter D layout -> row-major LDS tile
  _Float16* lt = lds[wave];
#pragma unroll
  for (int t = 0; t < 4; ++t)
#pragma unroll
    for (int r = 0; r < 8; ++r) {
      float v = acc[t][r];
      v = v >= 0.0f ? v : 0.2f * v;
      lt[(half * 8 + r) * CH + t * 16 + m] = (_Float16)v;
    }
  asm volatile("s_wait_dscnt 0" ::: "memory");  // same-wave LDS RAW fence

  // dense 64x64: 2 K-chunks from the LDS tile
  v8f acc2[4] = {};
#pragma unroll
  for (int c = 0; c < 2; ++c) {
    v16h a = make_a(lt + m * CH + c * 32 + half * 8);
    const _Float16* bp = pw2 + ((size_t)(c * 4) * 32 + lane) * 16;
#pragma unroll
    for (int t = 0; t < 4; ++t) {
      v16h b = *(const v16h*)(bp + (size_t)t * 32 * 16);
      acc2[t] = __builtin_amdgcn_wmma_f32_16x16x32_f16(
          false, a, false, b, (short)0, acc2[t], false, false);
    }
  }
#pragma unroll
  for (int t = 0; t < 4; ++t)
#pragma unroll
    for (int r = 0; r < 8; ++r) {
      float v = acc2[t][r];
      v = v >= 0.0f ? v : 0.2f * v;
      out[(rowBase + half * 8 + r) * CH + t * 16 + m] = (_Float16)v;
    }
}

// Stage 2: raw sparse_conv(feat, W3) -> f32 (no activation)
__global__ __launch_bounds__(WPB * 32) void k_sparse_out(
    const _Float16* __restrict__ feat, const int* __restrict__ nbr,
    const _Float16* __restrict__ pw, float* __restrict__ out) {
  const int tid = threadIdx.x;
  const int wave = tid >> 5, lane = tid & 31;
  const int m = lane & 15, half = lane >> 4;
  const long rowBase = ((long)blockIdx.x * WPB + wave) * 16;

  v8f acc[4] = {};
  sparse_accum(feat, nbr, pw, rowBase, lane, acc);
#pragma unroll
  for (int t = 0; t < 4; ++t)
#pragma unroll
    for (int r = 0; r < 8; ++r)
      out[(rowBase + half * 8 + r) * CH + t * 16 + m] = acc[t][r];
}

// Repack f32 weight (chunks*32 x 64 row-major) into wave32 B-fragment layout:
// dst[((c*4 + t)*32 + lane)*16 + e] = W[c*32 + (lane>>4)*8 + (e<8 ? e : e+8)][t*16 + (lane&15)]
__global__ void k_repack(const float* __restrict__ src, _Float16* __restrict__ dst,
                         int chunks) {
  int i = blockIdx.x * blockDim.x + threadIdx.x;
  if (i >= chunks * 4 * 32) return;
  int l = i & 31, t = (i >> 5) & 3, c = i >> 7;
  int n = t * 16 + (l & 15);
  int kbase = c * 32 + (l >> 4) * 8;
  v16h tmp;
#pragma unroll
  for (int e = 0; e < 16; ++e) tmp[e] = (_Float16)src[(kbase + (e < 8 ? e : e + 8)) * CH + n];
  *(v16h*)(dst + (size_t)i * 16) = tmp;
}

// x2 half of input -> f16 gather source
__global__ void k_x2h(const float* __restrict__ x, _Float16* __restrict__ x2h) {
  long i = (long)blockIdx.x * blockDim.x + threadIdx.x;
  if (i >= (long)NPTS * CH) return;
  long row = i >> 6; int col = (int)(i & 63);
  x2h[i] = (_Float16)x[row * 128 + 64 + col];
}

// coupling: y = x[:, xoff:xoff+64] * exp(tanh(cg/2)) + ch ; write out half,
// optionally also emit f16 copy for the next bottleneck's gathers.
__global__ void k_couple(const float* __restrict__ x, const float* __restrict__ cg,
                         const float* __restrict__ ch, float* __restrict__ out,
                         _Float16* __restrict__ yh, int xoff) {
  long i = (long)blockIdx.x * blockDim.x + threadIdx.x;
  if (i >= (long)NPTS * CH) return;
  long row = i >> 6; int col = (int)(i & 63);
  float s = expf(tanhf(0.5f * cg[i]));  // exp(CLAMP*(2*sigmoid(v)-1)), CLAMP=1
  float y = x[row * 128 + xoff + col] * s + ch[i];
  out[row * 128 + xoff + col] = y;
  if (yh) yh[i] = (_Float16)y;
}

extern "C" void kernel_launch(void* const* d_in, const int* in_sizes, int n_in,
                              void* d_out, int out_size, void* d_ws, size_t ws_size,
                              hipStream_t stream) {
  const float* x   = (const float*)d_in[0];
  const int*   nbr = (const int*)d_in[1];
  const float* w[12];
  for (int i = 0; i < 12; ++i) w[i] = (const float*)d_in[2 + i];
  // d_in order: x, nbr, g1_w1,g1_w2,g1_w3, g2_w1,g2_w2,g2_w3,
  //             h1_w1,h1_w2,h1_w3, h2_w1,h2_w2,h2_w3
  float* out = (float*)d_out;

  char* p = (char*)d_ws;
  const size_t szH = (size_t)NPTS * CH * 2;   // f16 N x 64
  const size_t szF = (size_t)NPTS * CH * 4;   // f32 N x 64
  _Float16* x2h = (_Float16*)p; p += szH;
  _Float16* t2h = (_Float16*)p; p += szH;
  _Float16* y1h = (_Float16*)p; p += szH;
  float* cg = (float*)p; p += szF;
  float* ch = (float*)p; p += szF;
  _Float16* pk[12];
  for (int i = 0; i < 12; ++i) {
    int chunks = (i % 3 == 1) ? 2 : 54;       // w2 is 64x64, w1/w3 are 1728x64
    pk[i] = (_Float16*)p; p += (size_t)chunks * 4 * 32 * 16 * 2;
  }

  const long tot = (long)NPTS * CH;
  const int ewBlocks = (int)((tot + 255) / 256);

  k_x2h<<<ewBlocks, 256, 0, stream>>>(x, x2h);
  for (int i = 0; i < 12; ++i) {
    int chunks = (i % 3 == 1) ? 2 : 54;
    int total = chunks * 128;
    k_repack<<<(total + 255) / 256, 256, 0, stream>>>(w[i], pk[i], chunks);
  }

  // pk index map: g1:0,1,2  g2:3,4,5  h1:6,7,8  h2:9,10,11
  // --- first coupling: operates on x2 ---
  k_sparse_dense<<<CONV_BLOCKS, WPB * 32, 0, stream>>>(x2h, nbr, pk[3], pk[4], t2h);
  k_sparse_out  <<<CONV_BLOCKS, WPB * 32, 0, stream>>>(t2h, nbr, pk[5], cg);
  k_sparse_dense<<<CONV_BLOCKS, WPB * 32, 0, stream>>>(x2h, nbr, pk[9], pk[10], t2h);
  k_sparse_out  <<<CONV_BLOCKS, WPB * 32, 0, stream>>>(t2h, nbr, pk[11], ch);
  k_couple<<<ewBlocks, 256, 0, stream>>>(x, cg, ch, out, y1h, 0);

  // --- second coupling: operates on y1 ---
  k_sparse_dense<<<CONV_BLOCKS, WPB * 32, 0, stream>>>(y1h, nbr, pk[0], pk[1], t2h);
  k_sparse_out  <<<CONV_BLOCKS, WPB * 32, 0, stream>>>(t2h, nbr, pk[2], cg);
  k_sparse_dense<<<CONV_BLOCKS, WPB * 32, 0, stream>>>(y1h, nbr, pk[6], pk[7], t2h);
  k_sparse_out  <<<CONV_BLOCKS, WPB * 32, 0, stream>>>(t2h, nbr, pk[8], ch);
  k_couple<<<ewBlocks, 256, 0, stream>>>(x, cg, ch, out, (_Float16*)nullptr, 64);
}
